// SlotAttentionTransformer_37529424232851
// MI455X (gfx1250) — matrix operand order repaired and checked by reading, not compiled, not run on hardware
//
#include <hip/hip_runtime.h>
#include <hip/hip_bf16.h>

#ifndef __has_builtin
#define __has_builtin(x) 0
#endif

// ---------------------------------------------------------------------------
// Problem constants (from reference)
// ---------------------------------------------------------------------------
#define Bn     2
#define Wg     64
#define Nn     4096          // H*W
#define CINc   256
#define Dd     256           // DSLOT
#define Kslots 8
#define KFg    7
#define CDc    5
#define CCol   8
#define NITER  4
#define EPSc   1e-8f
#define SCALEc 0.0625f       // 256^-0.5
#define MOMc   0.5f

typedef __attribute__((ext_vector_type(16))) _Float16 v16h;
typedef __attribute__((ext_vector_type(8)))  float    v8f;
typedef __attribute__((ext_vector_type(4)))  float    v4f;
typedef __attribute__((ext_vector_type(4)))  int      v4i;

union H16 { v4f f4[2]; v16h v; };

// ---------------------------------------------------------------------------
// CDNA5 async global->LDS staging (guarded; ds-store fallback)
// Builtin signature (from clang diagnostic): param0 = int4 addrspace(1)*,
// then LDS pointer, imm offset, imm cpol.
// ---------------------------------------------------------------------------
#if __has_builtin(__builtin_amdgcn_global_load_async_to_lds_b128)
#define HAVE_ASYNC_LDS 1
#else
#define HAVE_ASYNC_LDS 0
#endif

typedef __attribute__((address_space(1))) v4i gv4i_t;
typedef __attribute__((address_space(3))) v4i lv4i_t;

__device__ __forceinline__ void stage16(const _Float16* g, _Float16* l) {
#if HAVE_ASYNC_LDS
  __builtin_amdgcn_global_load_async_to_lds_b128((gv4i_t*)g, (lv4i_t*)l, 0, 0);
#else
  *(v4f*)l = *(const v4f*)g;
#endif
}

template <short NW>
__device__ __forceinline__ void wait_async_le() {
#if HAVE_ASYNC_LDS
#if __has_builtin(__builtin_amdgcn_s_wait_asynccnt)
  __builtin_amdgcn_s_wait_asynccnt(NW);
#else
  asm volatile("s_wait_asynccnt %0" :: "i"(NW) : "memory");
#endif
#endif
}

__device__ __forceinline__ float redsum256(float* red, int t, float v) {
  red[t] = v; __syncthreads();
  #pragma unroll
  for (int s = 128; s > 0; s >>= 1) { if (t < s) red[t] += red[t + s]; __syncthreads(); }
  float r = red[0]; __syncthreads();
  return r;
}

__device__ __forceinline__ void gridxy(int n, float& gx, float& gy) {
  int x = n & (Wg - 1), y = n >> 6;
  gx = -1.f + 2.f * (float)x / 63.f;
  gy = -1.f + 2.f * (float)y / 63.f;
}

// ---------------------------------------------------------------------------
// Small-M WMMA GEMM: C[M,N] = A[M,K]*B[K,N], BT[N,K] given transposed.
// One wave per 16x16 tile, direct global loads. Used for M==16 GEMMs.
// ---------------------------------------------------------------------------
__global__ void __launch_bounds__(32) k_wmma_gemm_nt(
    const _Float16* __restrict__ A, int lda, long long sA,
    const _Float16* __restrict__ BT, int ldb, long long sB,
    float* __restrict__ C, int ldc, long long sC, int Kd)
{
  const _Float16* Ab = A  + (long long)blockIdx.z * sA;
  const _Float16* Bb = BT + (long long)blockIdx.z * sB;
  float*          Cb = C  + (long long)blockIdx.z * sC;
  int lane = threadIdx.x;
  int r  = lane & 15;
  int kh = lane >> 4;   // which K-half this lane holds
  const _Float16* Ar = Ab + (long long)(blockIdx.y * 16 + r) * lda;
  const _Float16* Br = Bb + (long long)(blockIdx.x * 16 + r) * ldb;
  v8f acc = {};
  for (int k0 = 0; k0 < Kd; k0 += 32) {
    __builtin_prefetch(Br + k0 + 256, 0, 0);   // global_prefetch_b8
    H16 fa, fb;
    fa.f4[0] = *(const v4f*)(Ar + k0 + kh * 8);        // K = k0 + kh*8 .. +7
    fa.f4[1] = *(const v4f*)(Ar + k0 + 16 + kh * 8);   // K = k0+16+kh*8 .. +7
    fb.f4[0] = *(const v4f*)(Br + k0 + kh * 8);        // K = k0 + kh*8 .. +7
    fb.f4[1] = *(const v4f*)(Br + k0 + 16 + kh * 8);   // K = k0+16+kh*8 .. +7
    acc = __builtin_amdgcn_wmma_f32_16x16x32_f16(false, fa.v, false, fb.v,
                                                 (short)0, acc, false, false);
  }
  int col  = blockIdx.x * 16 + r;
  int row0 = blockIdx.y * 16 + kh * 8;
  #pragma unroll
  for (int i = 0; i < 8; i++)
    Cb[(long long)(row0 + i) * ldc + col] = acc[i];
}

// ---------------------------------------------------------------------------
// Large WMMA GEMM: 4 waves/block; block computes 64x64 tile; wave w owns the
// 16-row strip (m = by*64 + w*16) x 64 cols. The shared 64x32-half B slab is
// staged in LDS via async global->LDS loads, double-buffered:
//   issue(next); s_wait_asynccnt<=2; barrier; 4x wmma from LDS; barrier
// A fragment is reused across the 4 WMMAs (4x fewer A fetches; B fetched once
// per block instead of once per wave).  Requires M%64==0, N%64==0, K%32==0.
// ---------------------------------------------------------------------------
__global__ void __launch_bounds__(128) k_wmma_gemm_nt_lds(
    const _Float16* __restrict__ A, int lda, long long sA,
    const _Float16* __restrict__ BT, int ldb, long long sB,
    float* __restrict__ C, int ldc, long long sC, int Kd)
{
  __shared__ _Float16 Bs[2][64 * 32];    // [buf][row 0..63][32 halves] = 2x4KB
  const _Float16* Ab = A  + (long long)blockIdx.z * sA;
  const _Float16* Bb = BT + (long long)blockIdx.z * sB;
  float*          Cb = C  + (long long)blockIdx.z * sC;
  int t = threadIdx.x;
  int wave = t >> 5, lane = t & 31;
  int r = lane & 15, kh = lane >> 4;
  int nb = blockIdx.x * 64;
  const _Float16* Ar = Ab + (long long)(blockIdx.y * 64 + wave * 16 + r) * lda;

  // staging role of this thread: one (row, 16-half segment) pair
  int srow = t & 63, seg = t >> 6;                        // seg 0/1
  const _Float16* Bg = Bb + (long long)(nb + srow) * ldb + seg * 16;
  _Float16* Bl0 = &Bs[0][srow * 32 + seg * 16];
  _Float16* Bl1 = &Bs[1][srow * 32 + seg * 16];

  v8f acc[4] = {};
  int nk = Kd / 32;

  // prologue: stage k-step 0 into buffer 0
  stage16(Bg, Bl0);
  stage16(Bg + 8, Bl0 + 8);

  for (int ks = 0; ks < nk; ks++) {
    int cur = ks & 1;
    int k0 = ks * 32;
    bool more = (ks + 1 < nk);
    if (more) {
      _Float16* ln = cur ? Bl0 : Bl1;
      const _Float16* gn = Bg + (k0 + 32);
      stage16(gn, ln);
      stage16(gn + 8, ln + 8);
      wait_async_le<2>();      // batch for `cur` (issued last step) is complete
    } else {
      wait_async_le<0>();
    }
    __syncthreads();           // every wave finished waiting on its own loads

    __builtin_prefetch(Ar + k0 + 128, 0, 0);   // stream A ahead
    H16 fa;
    fa.f4[0] = *(const v4f*)(Ar + k0 + kh * 16);
    fa.f4[1] = *(const v4f*)(Ar + k0 + kh * 16 + 8);
    #pragma unroll
    for (int j = 0; j < 4; j++) {
      H16 fb;
      const _Float16* bp = &Bs[cur][(j * 16 + r) * 32 + kh * 16];
      fb.f4[0] = *(const v4f*)(bp);
      fb.f4[1] = *(const v4f*)(bp + 8);
      acc[j] = __builtin_amdgcn_wmma_f32_16x16x32_f16(false, fa.v, false, fb.v,
                                                      (short)0, acc[j], false, false);
    }
    __syncthreads();           // all reads of buf `cur` done before re-staging it
  }

  int row0 = blockIdx.y * 64 + wave * 16 + kh * 8;
  #pragma unroll
  for (int j = 0; j < 4; j++)
    #pragma unroll
    for (int i = 0; i < 8; i++)
      Cb[(long long)(row0 + i) * ldc + nb + j * 16 + r] = acc[j][i];
}

// ---------------------------------------------------------------------------
// Elementwise converters
// ---------------------------------------------------------------------------
__global__ void k_cvt_f16(const float* __restrict__ src, _Float16* __restrict__ dst, int n) {
  int i = blockIdx.x * 256 + threadIdx.x;
  if (i < n) dst[i] = (_Float16)src[i];
}
__global__ void k_cvt_t_f16(const float* __restrict__ src, _Float16* __restrict__ dst,
                            int R, int C) {
  int i = blockIdx.x * 256 + threadIdx.x;
  if (i < R * C) { int r = i / C, c = i % C; dst[(long long)c * R + r] = (_Float16)src[i]; }
}

// ---------------------------------------------------------------------------
// LayerNorm of feat (rows of 256) -> f16
// ---------------------------------------------------------------------------
__global__ void k_ln_feat(const float* __restrict__ feat, const float* __restrict__ g,
                          const float* __restrict__ bb, _Float16* __restrict__ out) {
  __shared__ float red[256];
  int row = blockIdx.x, t = threadIdx.x;
  float x = feat[(size_t)row * CINc + t];
  float mean = redsum256(red, t, x) * (1.f / CINc);
  float d = x - mean;
  float var = redsum256(red, t, d * d) * (1.f / CINc);
  out[(size_t)row * CINc + t] = (_Float16)(d * rsqrtf(var + 1e-5f) * g[t] + bb[t]);
}

// LayerNorm of feat_color (rows of 8) -> f32, one row per thread
__global__ void k_ln_color(const float* __restrict__ fc, const float* __restrict__ g,
                           const float* __restrict__ bb, float* __restrict__ out) {
  int row = blockIdx.x * 256 + threadIdx.x;
  if (row >= Bn * Nn) return;
  float v[CCol]; float m = 0.f;
  #pragma unroll
  for (int i = 0; i < CCol; i++) { v[i] = fc[(size_t)row * CCol + i]; m += v[i]; }
  m *= (1.f / CCol);
  float var = 0.f;
  #pragma unroll
  for (int i = 0; i < CCol; i++) { float d = v[i] - m; var += d * d; }
  var *= (1.f / CCol);
  float inv = rsqrtf(var + 1e-5f);
  #pragma unroll
  for (int i = 0; i < CCol; i++)
    out[(size_t)row * CCol + i] = (v[i] - m) * inv * g[i] + bb[i];
}

// ---------------------------------------------------------------------------
// a = (ge_w[0]-ge_w[2])@mlp, b = (ge_w[1]-ge_w[3])@mlp, c = ge_b@mlp
// ---------------------------------------------------------------------------
__global__ void k_calc_abc(const float* __restrict__ ge_w, const float* __restrict__ ge_b,
                           const float* __restrict__ mlp,
                           float* __restrict__ av, float* __restrict__ bv, float* __restrict__ cv) {
  int d = threadIdx.x;
  float sa = 0.f, sb = 0.f, sc = 0.f;
  for (int j = 0; j < CINc; j++) {
    float m = mlp[(size_t)j * Dd + d];
    sa += (ge_w[0 * CINc + j] - ge_w[2 * CINc + j]) * m;
    sb += (ge_w[1 * CINc + j] - ge_w[3 * CINc + j]) * m;
    sc += ge_b[j] * m;
  }
  av[d] = sa; bv[d] = sb; cv[d] = sc;
}

// buf[b,n,d] += gx*a[d] + gy*b[d] + c[d]
__global__ void k_grid_add(float* __restrict__ buf, const float* __restrict__ av,
                           const float* __restrict__ bv, const float* __restrict__ cv) {
  int row = blockIdx.x, t = threadIdx.x;
  float gx, gy; gridxy(row & (Nn - 1), gx, gy);
  buf[(size_t)row * Dd + t] += gx * av[t] + gy * bv[t] + cv[t];
}

// broadcast initial state over batch
__global__ void k_init_state(const float* __restrict__ fg_pos_init,
                             const float* __restrict__ slots_fg,
                             const float* __restrict__ slots_bg,
                             float* __restrict__ fg_pos, float* __restrict__ slot_fg,
                             float* __restrict__ slot_bg) {
  int t = blockIdx.x * 256 + threadIdx.x;
  if (t < Bn * KFg * Dd) slot_fg[t] = slots_fg[t % (KFg * Dd)];
  if (t < Bn * Dd)       slot_bg[t] = slots_bg[t % Dd];
  if (t < Bn * KFg * 2)  fg_pos[t]  = fg_pos_init[t % (KFg * 2)];
}

// ---------------------------------------------------------------------------
// adaLN + q projection. block (s, b): s=0 -> bg slot, s=1..7 -> fg slot s-1.
// Writes f16 rows into padded (16 x 256) Q matrices; also qa=q.a, qb=q.b (fg).
// ---------------------------------------------------------------------------
__global__ void k_qproj(const float* __restrict__ slot_fg, const float* __restrict__ slot_bg,
    const float* __restrict__ cam,
    const float* gfg, const float* bfg2, const float* cwfg, const float* cbfg,
    const float* gbg, const float* bbg2, const float* cwbg, const float* cbbg,
    const float* qfw, const float* qbw,
    const float* __restrict__ a_fg, const float* __restrict__ b_fg,
    _Float16* __restrict__ Qfg, _Float16* __restrict__ Qbg,
    float* __restrict__ qa, float* __restrict__ qb)
{
  __shared__ float red[256];
  __shared__ float h[256];
  int s = blockIdx.x, bb = blockIdx.y, t = threadIdx.x;
  bool isbg = (s == 0);
  const float* x  = isbg ? slot_bg + (size_t)bb * Dd : slot_fg + (size_t)(bb * KFg + s - 1) * Dd;
  const float* g  = isbg ? gbg  : gfg;
  const float* be = isbg ? bbg2 : bfg2;
  const float* cw = isbg ? cwbg : cwfg;
  const float* cb = isbg ? cbbg : cbfg;
  const float* qw = isbg ? qbw  : qfw;
  float v = x[t];
  float mean = redsum256(red, t, v) * (1.f / Dd);
  float d = v - mean;
  float var = redsum256(red, t, d * d) * (1.f / Dd);
  float xn = d * rsqrtf(var + 1e-5f) * g[t] + be[t];
  float gacc = cb[t], bacc = cb[Dd + t];
  for (int j = 0; j < CDc; j++) {
    float c = cam[bb * CDc + j];
    gacc += c * cw[(size_t)j * 2 * Dd + t];
    bacc += c * cw[(size_t)j * 2 * Dd + Dd + t];
  }
  float hv = xn * (1.f + tanhf(gacc)) + tanhf(bacc);
  h[t] = hv; __syncthreads();
  float q = 0.f;
  for (int c = 0; c < Dd; c++) q += h[c] * qw[(size_t)c * Dd + t];
  if (isbg) {
    Qbg[((size_t)bb * 16) * Dd + t] = (_Float16)q;
  } else {
    Qfg[((size_t)(bb * 16 + s - 1)) * Dd + t] = (_Float16)q;
    float ra = redsum256(red, t, q * a_fg[t]);
    float rb = redsum256(red, t, q * b_fg[t]);
    if (t == 0) { qa[bb * KFg + s - 1] = ra; qb[bb * KFg + s - 1] = rb; }
  }
}

// ---------------------------------------------------------------------------
// Softmax over slot axis (8) per pixel, +EPS; writes attn to d_out region and
// accumulates sum over N per (b,slot).
// ---------------------------------------------------------------------------
__global__ void k_attn_softmax(const float* __restrict__ Lfg, const float* __restrict__ Lbg,
                               const float* __restrict__ qa, const float* __restrict__ qb,
                               const float* __restrict__ fg_pos,
                               float* __restrict__ attn_out, float* __restrict__ sum_attn) {
  __shared__ float ssum[Kslots];
  int t = threadIdx.x;
  if (t < Kslots) ssum[t] = 0.f;
  __syncthreads();
  int idx = blockIdx.x * 256 + t;
  int bb = idx >> 12, n = idx & (Nn - 1);
  float lg[Kslots];
  lg[0] = Lbg[((size_t)bb * 16) * Nn + n] * SCALEc;
  #pragma unroll
  for (int k = 0; k < KFg; k++) {
    float px = fg_pos[(bb * KFg + k) * 2 + 0];
    float py = fg_pos[(bb * KFg + k) * 2 + 1];
    lg[k + 1] = (Lfg[((size_t)(bb * 16 + k)) * Nn + n]
                 - px * qa[bb * KFg + k] - py * qb[bb * KFg + k]) * SCALEc;
  }
  float mx = lg[0];
  #pragma unroll
  for (int k = 1; k < Kslots; k++) mx = fmaxf(mx, lg[k]);
  float ssm = 0.f;
  #pragma unroll
  for (int k = 0; k < Kslots; k++) { lg[k] = expf(lg[k] - mx); ssm += lg[k]; }
  #pragma unroll
  for (int k = 0; k < Kslots; k++) {
    float av = lg[k] / ssm + EPSc;
    attn_out[((size_t)(bb * Kslots + k)) * Nn + n] = av;
    atomicAdd(&ssum[k], av);
  }
  __syncthreads();
  if (t < Kslots) atomicAdd(&sum_attn[bb * Kslots + t], ssum[t]);
}

// normalize over N, write padded f16 W rows, accumulate S_gx/S_gy (fg)
__global__ void k_attn_norm(const float* __restrict__ attn, const float* __restrict__ sum_attn,
                            _Float16* __restrict__ Wfg, _Float16* __restrict__ Wbg,
                            float* __restrict__ sgx, float* __restrict__ sgy) {
  __shared__ float sx[KFg], sy[KFg];
  int t = threadIdx.x;
  if (t < KFg) { sx[t] = 0.f; sy[t] = 0.f; }
  __syncthreads();
  int idx = blockIdx.x * 256 + t;
  int bb = idx >> 12, n = idx & (Nn - 1);
  float gx, gy; gridxy(n, gx, gy);
  float wb = attn[((size_t)bb * Kslots) * Nn + n] / sum_attn[bb * Kslots];
  Wbg[((size_t)bb * 16) * Nn + n] = (_Float16)wb;
  #pragma unroll
  for (int k = 0; k < KFg; k++) {
    float wv = attn[((size_t)(bb * Kslots + 1 + k)) * Nn + n] / sum_attn[bb * Kslots + 1 + k];
    Wfg[((size_t)(bb * 16 + k)) * Nn + n] = (_Float16)wv;
    atomicAdd(&sx[k], wv * gx);
    atomicAdd(&sy[k], wv * gy);
  }
  __syncthreads();
  if (t < KFg) { atomicAdd(&sgx[bb * KFg + t], sx[t]); atomicAdd(&sgy[bb * KFg + t], sy[t]); }
}

// slot += update ( - rank-1 pos correction for fg );  pos momentum update
__global__ void k_apply_updates(float* __restrict__ slot_fg, float* __restrict__ slot_bg,
                                float* __restrict__ fg_pos,
                                const float* __restrict__ Ufg, const float* __restrict__ Ubg,
                                const float* __restrict__ av, const float* __restrict__ bv,
                                const float* __restrict__ sgx, const float* __restrict__ sgy) {
  int s = blockIdx.x, bb = blockIdx.y, t = threadIdx.x;
  if (s == 0) {
    slot_bg[(size_t)bb * Dd + t] += Ubg[((size_t)bb * 16) * Dd + t];
  } else {
    int k = s - 1;
    float px = fg_pos[(bb * KFg + k) * 2 + 0];
    float py = fg_pos[(bb * KFg + k) * 2 + 1];
    __syncthreads();   // everyone reads old pos before lanes 0/1 overwrite it
    slot_fg[(size_t)(bb * KFg + k) * Dd + t] +=
        Ufg[((size_t)(bb * 16 + k)) * Dd + t] - px * av[t] - py * bv[t];
    if (t == 0) fg_pos[(bb * KFg + k) * 2 + 0] = sgx[bb * KFg + k] * (1.f - MOMc) + px * MOMc;
    if (t == 1) fg_pos[(bb * KFg + k) * 2 + 1] = sgy[bb * KFg + k] * (1.f - MOMc) + py * MOMc;
  }
}

// adaLN + gelu MLP residual on slots
__global__ void k_slot_mlp(float* __restrict__ slot_fg, float* __restrict__ slot_bg,
    const float* __restrict__ cam,
    const float* gfg, const float* bfg2, const float* cwfg, const float* cbfg,
    const float* gbg, const float* bbg2, const float* cwbg, const float* cbbg,
    const float* w1f, const float* b1f, const float* w2f, const float* b2f,
    const float* w1b, const float* b1b, const float* w2b, const float* b2b)
{
  __shared__ float red[256];
  __shared__ float h[256];
  int s = blockIdx.x, bb = blockIdx.y, t = threadIdx.x;
  bool isbg = (s == 0);
  float* slot = isbg ? slot_bg + (size_t)bb * Dd : slot_fg + (size_t)(bb * KFg + s - 1) * Dd;
  const float* g  = isbg ? gbg  : gfg;
  const float* be = isbg ? bbg2 : bfg2;
  const float* cw = isbg ? cwbg : cwfg;
  const float* cb = isbg ? cbbg : cbfg;
  const float* w1 = isbg ? w1b : w1f; const float* b1 = isbg ? b1b : b1f;
  const float* w2 = isbg ? w2b : w2f; const float* b2 = isbg ? b2b : b2f;
  float v = slot[t];
  float mean = redsum256(red, t, v) * (1.f / Dd);
  float d = v - mean;
  float var = redsum256(red, t, d * d) * (1.f / Dd);
  float xn = d * rsqrtf(var + 1e-5f) * g[t] + be[t];
  float gacc = cb[t], bacc = cb[Dd + t];
  for (int j = 0; j < CDc; j++) {
    float c = cam[bb * CDc + j];
    gacc += c * cw[(size_t)j * 2 * Dd + t];
    bacc += c * cw[(size_t)j * 2 * Dd + Dd + t];
  }
  h[t] = xn * (1.f + tanhf(gacc)) + tanhf(bacc);
  __syncthreads();
  float acc = b1[t];
  for (int c = 0; c < Dd; c++) acc += h[c] * w1[(size_t)c * Dd + t];
  float ge = 0.5f * acc * (1.f + erff(acc * 0.70710678118f));   // exact gelu
  __syncthreads();
  h[t] = ge;
  __syncthreads();
  float acc2 = b2[t];
  for (int c = 0; c < Dd; c++) acc2 += h[c] * w2[(size_t)c * Dd + t];
  slot[t] = v + acc2;
}

// final iteration: colors, final pos (momentum + tanh term), assemble outputs
__global__ void k_finalize(const float* __restrict__ slot_fg, const float* __restrict__ slot_bg,
                           const float* __restrict__ attn, const float* __restrict__ sum_attn,
                           const float* __restrict__ fcn, const float* __restrict__ pos_w,
                           const float* __restrict__ pos_b, const float* __restrict__ fg_pos,
                           const float* __restrict__ sgx, const float* __restrict__ sgy,
                           float* __restrict__ out) {
  __shared__ float red[256];
  int s = blockIdx.x, bb = blockIdx.y, t = threadIdx.x;
  float col[CCol]; float pw0 = 0.f, pw1 = 0.f;
  #pragma unroll
  for (int c = 0; c < CCol; c++) col[c] = 0.f;
  float ssum = sum_attn[bb * Kslots + s];
  for (int n = t; n < Nn; n += 256) {
    float wv = attn[((size_t)(bb * Kslots + s)) * Nn + n] / ssum;
    #pragma unroll
    for (int c = 0; c < CCol; c++) col[c] += wv * fcn[((size_t)(bb * Nn + n)) * CCol + c];
    if (s > 0) { pw0 += wv * pos_w[n * 2 + 0]; pw1 += wv * pos_w[n * 2 + 1]; }
  }
  float* orow = out + (size_t)(bb * Kslots + s) * (Dd + CCol);
  orow[t] = (s == 0) ? slot_bg[(size_t)bb * Dd + t]
                     : slot_fg[(size_t)(bb * KFg + s - 1) * Dd + t];
  #pragma unroll
  for (int c = 0; c < CCol; c++) {
    float r = redsum256(red, t, col[c]);
    if (t == 0) orow[Dd + c] = r;
  }
  if (s > 0) {
    float r0 = redsum256(red, t, pw0);
    float r1 = redsum256(red, t, pw1);
    if (t == 0) {
      int k = s - 1;
      float px = fg_pos[(bb * KFg + k) * 2 + 0];
      float py = fg_pos[(bb * KFg + k) * 2 + 1];
      float nx = sgx[bb * KFg + k] * (1.f - MOMc) + px * MOMc + tanhf(r0 + pos_b[0]) * 0.1f;
      float ny = sgy[bb * KFg + k] * (1.f - MOMc) + py * MOMc + tanhf(r1 + pos_b[1]) * 0.1f;
      float* op = out + Bn * Kslots * (Dd + CCol) + Bn * Kslots * Nn + (bb * KFg + k) * 2;
      op[0] = nx; op[1] = ny;
    }
  }
}

// ---------------------------------------------------------------------------
// Host driver
// ---------------------------------------------------------------------------
extern "C" void kernel_launch(void* const* d_in, const int* in_sizes, int n_in,
                              void* d_out, int out_size, void* d_ws, size_t ws_size,
                              hipStream_t stream) {
  (void)in_sizes; (void)n_in; (void)out_size; (void)ws_size;
  // setup_inputs() flattening order (params dict depth-first, insertion order)
  const float* feat        = (const float*)d_in[0];
  const float* cam         = (const float*)d_in[1];
  const float* feat_color  = (const float*)d_in[2];
  const float* fg_pos_init = (const float*)d_in[3];
  const float* slots_fg0   = (const float*)d_in[4];
  const float* slots_bg0   = (const float*)d_in[5];
  const float* ge_w        = (const float*)d_in[6];
  const float* ge_b        = (const float*)d_in[7];
  const float* k_fg        = (const float*)d_in[8];
  const float* v_fg        = (const float*)d_in[9];
  const float* k_bg        = (const float*)d_in[10];
  const float* v_bg        = (const float*)d_in[11];
  const float* mlp_fg      = (const float*)d_in[12];
  const float* mlp_bg      = (const float*)d_in[13];
  const float* q_fg_w      = (const float*)d_in[14];
  const float* q_bg_w      = (const float*)d_in[15];
  const float* qfg_g  = (const float*)d_in[16]; const float* qfg_b  = (const float*)d_in[17];
  const float* qfg_cw = (const float*)d_in[18]; const float* qfg_cb = (const float*)d_in[19];
  const float* qbg_g  = (const float*)d_in[20]; const float* qbg_b  = (const float*)d_in[21];
  const float* qbg_cw = (const float*)d_in[22]; const float* qbg_cb = (const float*)d_in[23];
  const float* mfg_g  = (const float*)d_in[24]; const float* mfg_b  = (const float*)d_in[25];
  const float* mfg_cw = (const float*)d_in[26]; const float* mfg_cb = (const float*)d_in[27];
  const float* mbg_g  = (const float*)d_in[28]; const float* mbg_b  = (const float*)d_in[29];
  const float* mbg_cw = (const float*)d_in[30]; const float* mbg_cb = (const float*)d_in[31];
  const float* nf_g   = (const float*)d_in[32]; const float* nf_b   = (const float*)d_in[33];
  const float* nc_g   = (const float*)d_in[34]; const float* nc_b   = (const float*)d_in[35];
  const float* mfg_w1 = (const float*)d_in[36]; const float* mfg_b1 = (const float*)d_in[37];
  const float* mfg_w2 = (const float*)d_in[38]; const float* mfg_b2 = (const float*)d_in[39];
  const float* mbg_w1 = (const float*)d_in[40]; const float* mbg_b1 = (const float*)d_in[41];
  const float* mbg_w2 = (const float*)d_in[42]; const float* mbg_b2 = (const float*)d_in[43];
  const float* pos_w  = (const float*)d_in[44]; const float* pos_b  = (const float*)d_in[45];

  float* out = (float*)d_out;
  float* attn_out = out + Bn * Kslots * (Dd + CCol);   // [B,8,N] region of d_out

  // -------- workspace bump allocator (256-B aligned) ----------------------
  char* base = (char*)d_ws; size_t off = 0;
  auto alloc = [&](size_t bytes) -> void* {
    void* p = base + off; off = (off + bytes + 255) & ~(size_t)255; return p;
  };
  _Float16* featf_h = (_Float16*)alloc((size_t)Bn * Nn * CINc * 2);
  float*    fcn     = (float*)   alloc((size_t)Bn * Nn * CCol * 4);
  _Float16* kfg_h   = (_Float16*)alloc(65536 * 2);
  _Float16* vfg_h   = (_Float16*)alloc(65536 * 2);
  _Float16* kbg_h   = (_Float16*)alloc(65536 * 2);
  _Float16* vbg_h   = (_Float16*)alloc(65536 * 2);
  _Float16* mlpfT_h = (_Float16*)alloc(65536 * 2);
  _Float16* mlpbT_h = (_Float16*)alloc(65536 * 2);
  float* kw  = (float*)alloc(65536 * 4); float* vw  = (float*)alloc(65536 * 4);
  float* kbw = (float*)alloc(65536 * 4); float* vbw = (float*)alloc(65536 * 4);
  _Float16* kwT  = (_Float16*)alloc(65536 * 2); _Float16* vwT  = (_Float16*)alloc(65536 * 2);
  _Float16* kbwT = (_Float16*)alloc(65536 * 2); _Float16* vbwT = (_Float16*)alloc(65536 * 2);
  float* KF0 = (float*)alloc((size_t)Bn * Nn * Dd * 4);
  float* VF0 = (float*)alloc((size_t)Bn * Nn * Dd * 4);
  float* KB  = (float*)alloc((size_t)Bn * Nn * Dd * 4);
  float* VB  = (float*)alloc((size_t)Bn * Nn * Dd * 4);
  _Float16* KF0h = (_Float16*)alloc((size_t)Bn * Nn * Dd * 2);
  _Float16* KBh  = (_Float16*)alloc((size_t)Bn * Nn * Dd * 2);
  _Float16* VF0t = (_Float16*)alloc((size_t)Bn * Dd * Nn * 2);
  _Float16* VBt  = (_Float16*)alloc((size_t)Bn * Dd * Nn * 2);
  float* afg = (float*)alloc(256 * 4); float* bfg = (float*)alloc(256 * 4);
  float* cfg = (float*)alloc(256 * 4);
  float* abg = (float*)alloc(256 * 4); float* bbg = (float*)alloc(256 * 4);
  float* cbg = (float*)alloc(256 * 4);
  float* slot_fg = (float*)alloc((size_t)Bn * KFg * Dd * 4);
  float* slot_bg = (float*)alloc((size_t)Bn * Dd * 4);
  float* fg_pos  = (float*)alloc(Bn * KFg * 2 * 4);
  _Float16* Qfg_h = (_Float16*)alloc((size_t)Bn * 16 * Dd * 2);
  _Float16* Qbg_h = (_Float16*)alloc((size_t)Bn * 16 * Dd * 2);
  float* qa = (float*)alloc(Bn * KFg * 4); float* qb = (float*)alloc(Bn * KFg * 4);
  float* Lfg = (float*)alloc((size_t)Bn * 16 * Nn * 4);
  float* Lbg = (float*)alloc((size_t)Bn * 16 * Nn * 4);
  float* sum_attn = (float*)alloc(Bn * Kslots * 4);
  float* sgx = (float*)alloc(Bn * KFg * 4); float* sgy = (float*)alloc(Bn * KFg * 4);
  _Float16* Wfg_h = (_Float16*)alloc((size_t)Bn * 16 * Nn * 2);
  _Float16* Wbg_h = (_Float16*)alloc((size_t)Bn * 16 * Nn * 2);
  float* Ufg = (float*)alloc((size_t)Bn * 16 * Dd * 4);
  float* Ubg = (float*)alloc((size_t)Bn * 16 * Dd * 4);

  // small-M GEMM (one wave / 16x16 tile)
  auto gemm = [&](const _Float16* A, int lda, long long sA,
                  const _Float16* BT, int ldb, long long sB,
                  float* C, int ldc, long long sC, int M, int N, int Kd, int batch) {
    dim3 g(N / 16, M / 16, batch);
    k_wmma_gemm_nt<<<g, dim3(32), 0, stream>>>(A, lda, sA, BT, ldb, sB, C, ldc, sC, Kd);
  };
  // large GEMM (async-LDS staged, 64x64 tile / block)
  auto gemm64 = [&](const _Float16* A, int lda, long long sA,
                    const _Float16* BT, int ldb, long long sB,
                    float* C, int ldc, long long sC, int M, int N, int Kd, int batch) {
    dim3 g(N / 64, M / 64, batch);
    k_wmma_gemm_nt_lds<<<g, dim3(128), 0, stream>>>(A, lda, sA, BT, ldb, sB, C, ldc, sC, Kd);
  };

  // zero the padded operand rows once per call (graph-capturable)
  (void)hipMemsetAsync(Qfg_h, 0, (size_t)Bn * 16 * Dd * 2, stream);
  (void)hipMemsetAsync(Qbg_h, 0, (size_t)Bn * 16 * Dd * 2, stream);
  (void)hipMemsetAsync(Wfg_h, 0, (size_t)Bn * 16 * Nn * 2, stream);
  (void)hipMemsetAsync(Wbg_h, 0, (size_t)Bn * 16 * Nn * 2, stream);

  // ---- preprocessing -----------------------------------------------------
  k_ln_feat<<<Bn * Nn, 256, 0, stream>>>(feat, nf_g, nf_b, featf_h);
  k_ln_color<<<(Bn * Nn + 255) / 256, 256, 0, stream>>>(feat_color, nc_g, nc_b, fcn);

  k_cvt_f16<<<256, 256, 0, stream>>>(k_fg, kfg_h, 65536);
  k_cvt_f16<<<256, 256, 0, stream>>>(v_fg, vfg_h, 65536);
  k_cvt_f16<<<256, 256, 0, stream>>>(k_bg, kbg_h, 65536);
  k_cvt_f16<<<256, 256, 0, stream>>>(v_bg, vbg_h, 65536);
  k_cvt_t_f16<<<256, 256, 0, stream>>>(mlp_fg, mlpfT_h, 256, 256);
  k_cvt_t_f16<<<256, 256, 0, stream>>>(mlp_bg, mlpbT_h, 256, 256);

  // weight chains  kw = k_fg @ mlp_fg  etc.  (WMMA, LDS-staged)
  gemm64(kfg_h, 256, 0, mlpfT_h, 256, 0, kw,  256, 0, 256, 256, 256, 1);
  gemm64(vfg_h, 256, 0, mlpfT_h, 256, 0, vw,  256, 0, 256, 256, 256, 1);
  gemm64(kbg_h, 256, 0, mlpbT_h, 256, 0, kbw, 256, 0, 256, 256, 256, 1);
  gemm64(vbg_h, 256, 0, mlpbT_h, 256, 0, vbw, 256, 0, 256, 256, 256, 1);
  k_cvt_t_f16<<<256, 256, 0, stream>>>(kw,  kwT,  256, 256);
  k_cvt_t_f16<<<256, 256, 0, stream>>>(vw,  vwT,  256, 256);
  k_cvt_t_f16<<<256, 256, 0, stream>>>(kbw, kbwT, 256, 256);
  k_cvt_t_f16<<<256, 256, 0, stream>>>(vbw, vbwT, 256, 256);

  // big GEMMs: featf(8192x256) @ 256x256  (WMMA, async-LDS staged)
  gemm64(featf_h, 256, 0, kwT,  256, 0, KF0, 256, 0, Bn * Nn, 256, 256, 1);
  gemm64(featf_h, 256, 0, vwT,  256, 0, VF0, 256, 0, Bn * Nn, 256, 256, 1);
  gemm64(featf_h, 256, 0, kbwT, 256, 0, KB,  256, 0, Bn * Nn, 256, 256, 1);
  gemm64(featf_h, 256, 0, vbwT, 256, 0, VB,  256, 0, Bn * Nn, 256, 256, 1);

  // rank-1 grid terms
  k_calc_abc<<<1, 256, 0, stream>>>(ge_w, ge_b, mlp_fg, afg, bfg, cfg);
  k_calc_abc<<<1, 256, 0, stream>>>(ge_w, ge_b, mlp_bg, abg, bbg, cbg);
  k_grid_add<<<Bn * Nn, 256, 0, stream>>>(KF0, afg, bfg, cfg);
  k_grid_add<<<Bn * Nn, 256, 0, stream>>>(VF0, afg, bfg, cfg);
  k_grid_add<<<Bn * Nn, 256, 0, stream>>>(KB, abg, bbg, cbg);
  k_grid_add<<<Bn * Nn, 256, 0, stream>>>(VB, abg, bbg, cbg);

  // f16 copies used every iteration
  k_cvt_f16<<<(Bn * Nn * Dd + 255) / 256, 256, 0, stream>>>(KF0, KF0h, Bn * Nn * Dd);
  k_cvt_f16<<<(Bn * Nn * Dd + 255) / 256, 256, 0, stream>>>(KB,  KBh,  Bn * Nn * Dd);
  for (int b = 0; b < Bn; b++) {
    k_cvt_t_f16<<<(Nn * Dd + 255) / 256, 256, 0, stream>>>(VF0 + (size_t)b * Nn * Dd,
                                                           VF0t + (size_t)b * Dd * Nn, Nn, Dd);
    k_cvt_t_f16<<<(Nn * Dd + 255) / 256, 256, 0, stream>>>(VB + (size_t)b * Nn * Dd,
                                                           VBt + (size_t)b * Dd * Nn, Nn, Dd);
  }

  k_init_state<<<(Bn * KFg * Dd + 255) / 256, 256, 0, stream>>>(
      fg_pos_init, slots_fg0, slots_bg0, fg_pos, slot_fg, slot_bg);

  // ---- iterations --------------------------------------------------------
  for (int it = 0; it < NITER; it++) {
    k_qproj<<<dim3(Kslots, Bn), 256, 0, stream>>>(
        slot_fg, slot_bg, cam,
        qfg_g, qfg_b, qfg_cw, qfg_cb, qbg_g, qbg_b, qbg_cw, qbg_cb,
        q_fg_w, q_bg_w, afg, bfg, Qfg_h, Qbg_h, qa, qb);

    // logits: Q(16x256) @ K(4096x256)^T per batch  (WMMA)
    gemm(Qfg_h, 256, 16 * 256, KF0h, 256, (long long)Nn * Dd, Lfg, Nn, 16 * Nn,
         16, Nn, 256, Bn);
    gemm(Qbg_h, 256, 16 * 256, KBh, 256, (long long)Nn * Dd, Lbg, Nn, 16 * Nn,
         16, Nn, 256, Bn);

    (void)hipMemsetAsync(sum_attn, 0, Bn * Kslots * 4, stream);
    (void)hipMemsetAsync(sgx, 0, Bn * KFg * 4, stream);
    (void)hipMemsetAsync(sgy, 0, Bn * KFg * 4, stream);
    k_attn_softmax<<<(Bn * Nn) / 256, 256, 0, stream>>>(Lfg, Lbg, qa, qb, fg_pos,
                                                        attn_out, sum_attn);
    k_attn_norm<<<(Bn * Nn) / 256, 256, 0, stream>>>(attn_out, sum_attn,
                                                     Wfg_h, Wbg_h, sgx, sgy);
    if (it != NITER - 1) {
      // updates: W(16x4096) @ V(4096x256) per batch  (WMMA, K=4096)
      gemm(Wfg_h, Nn, 16 * Nn, VF0t, Nn, (long long)Dd * Nn, Ufg, Dd, 16 * Dd,
           16, Dd, Nn, Bn);
      gemm(Wbg_h, Nn, 16 * Nn, VBt, Nn, (long long)Dd * Nn, Ubg, Dd, 16 * Dd,
           16, Dd, Nn, Bn);
      k_apply_updates<<<dim3(Kslots, Bn), 256, 0, stream>>>(
          slot_fg, slot_bg, fg_pos, Ufg, Ubg, afg, bfg, sgx, sgy);
      k_slot_mlp<<<dim3(Kslots, Bn), 256, 0, stream>>>(
          slot_fg, slot_bg, cam,
          mfg_g, mfg_b, mfg_cw, mfg_cb, mbg_g, mbg_b, mbg_cw, mbg_cb,
          mfg_w1, mfg_b1, mfg_w2, mfg_b2, mbg_w1, mbg_b1, mbg_w2, mbg_b2);
    } else {
      k_finalize<<<dim3(Kslots, Bn), 256, 0, stream>>>(
          slot_fg, slot_bg, attn_out, sum_attn, fcn, pos_w, pos_b,
          fg_pos, sgx, sgy, out);
    }
  }
}
